// EquivariantScalar_42468636622963
// MI455X (gfx1250) — matrix operand order repaired and checked by reading, not compiled, not run on hardware
//
#include <hip/hip_runtime.h>
#include <math.h>

typedef __bf16 bf16_t;
typedef __attribute__((ext_vector_type(16))) __bf16 v16bf;
typedef __attribute__((ext_vector_type(8)))  __bf16 v8bf;
typedef __attribute__((ext_vector_type(8)))  float  v8f;

#define F_DIM   128
#define TM      32          // nodes per workgroup
#define NTHREADS 256        // 8 waves (wave32)

union Frag16 { v16bf v; v8bf h[2]; };

// A operand (16x32 bf16) from LDS tile [m][k] row-major (row stride rs).
// ISA layout: m = lane%16 ; element e -> K = (e/8)*16 + (lane/16)*8 + e%8
// => two contiguous 16-byte loads per lane.
__device__ __forceinline__ v16bf load_a_frag(const bf16_t* A, int rs, int mbase,
                                             int kbase, int lane) {
  const int m  = mbase + (lane & 15);
  const int ko = kbase + ((lane >> 4) << 3);
  const v8bf* p = (const v8bf*)(A + m * rs + ko);
  Frag16 f;
  f.h[0] = p[0];   // K = ko .. ko+7
  f.h[1] = p[2];   // K = ko+16 .. ko+23
  return f.v;
}

// B operand (32x16 bf16) straight from global weights stored [n][k] row-major
// (original layout), row stride kdim: n = lane%16, element e ->
// K = kbase + (e/8)*16 + (lane/16)*8 + e%8 => two 16-byte loads.
__device__ __forceinline__ v16bf load_b_frag(const bf16_t* __restrict__ W, int kdim,
                                             int nbase, int kbase, int lane) {
  const int n  = nbase + (lane & 15);
  const int kk = kbase + ((lane >> 4) << 3);
  const v8bf* p = (const v8bf*)(W + n * kdim + kk);
  Frag16 f;
  f.h[0] = p[0];
  f.h[1] = p[2];
  return f.v;
}

__device__ __forceinline__ v8f wmma_bf16(v16bf a, v16bf b, v8f c) {
  return __builtin_amdgcn_wmma_f32_16x16x32_bf16(false, a, false, b, (short)0, c,
                                                 false, false);
}

// D fragment (16x16 f32): element r -> M = mbase + r + (lane/16)*8, N = nbase + lane%16
__device__ __forceinline__ void store_d_bf16(bf16_t* T, int rs, int mbase, int nbase,
                                             int lane, v8f acc) {
  const int n  = nbase + (lane & 15);
  const int m0 = mbase + ((lane >> 4) << 3);
#pragma unroll
  for (int r = 0; r < 8; ++r) T[(m0 + r) * rs + n] = (bf16_t)acc[r];
}

// ---------------------------------------------------------------------------
__global__ void cvt_weights_bf16(const float* __restrict__ w1,
                                 const float* __restrict__ w2,
                                 const float* __restrict__ mw1,
                                 const float* __restrict__ mw2,
                                 bf16_t* __restrict__ out) {
  const int i = blockIdx.x * blockDim.x + threadIdx.x;
  float v;
  if      (i <  32768) v = w1[i];
  else if (i <  65536) v = w2[i - 32768];
  else if (i < 131072) v = mw1[i - 65536];
  else if (i < 196608) v = mw2[i - 131072];
  else return;
  out[i] = (bf16_t)v;
}

// ---------------------------------------------------------------------------
__global__ __launch_bounds__(NTHREADS)
void gated_equiv_fused(const float* __restrict__ sg,
                       const float* __restrict__ vg,
                       const bf16_t* __restrict__ wb,
                       const float* __restrict__ b1g,
                       const float* __restrict__ b2g,
                       const float* __restrict__ outw,
                       const float* __restrict__ outb,
                       float* __restrict__ sfin,
                       int nnodes) {
  __shared__ __align__(16) bf16_t vA   [3 * TM * F_DIM];
  __shared__ __align__(16) bf16_t vT   [3 * TM * F_DIM];
  __shared__ __align__(16) bf16_t sA   [TM * F_DIM];
  __shared__ __align__(16) bf16_t v2nB [TM * F_DIM];
  __shared__ __align__(16) bf16_t hA   [TM * F_DIM];
  __shared__ __align__(16) bf16_t gateB[TM * F_DIM];

  const int tid   = threadIdx.x;
  const int lane  = tid & 31;
  const int w     = tid >> 5;
  const int node0 = blockIdx.x * TM;
  const int nvalid = (nnodes - node0 < TM) ? (nnodes - node0) : TM;

  for (int idx = tid; idx < TM * F_DIM; idx += NTHREADS) {
    const int m = idx / F_DIM;
    sA[idx] = (bf16_t)((m < nvalid) ? sg[(size_t)node0 * F_DIM + idx] : 0.f);
  }
  for (int idx = tid; idx < 3 * TM * F_DIM; idx += NTHREADS) {
    const int m = idx / (3 * F_DIM);
    vA[idx] = (bf16_t)((m < nvalid) ? vg[(size_t)node0 * 3 * F_DIM + idx] : 0.f);
  }
  __syncthreads();

  const bf16_t* w1b  = wb;
  const bf16_t* w2b  = wb + 32768;
  const bf16_t* mw1b = wb + 65536;
  const bf16_t* mw2b = wb + 131072;

  for (int blk = 0; blk < 2; ++blk) {
    const bf16_t* W1 = w1b  + blk * F_DIM * F_DIM;
    const bf16_t* W2 = w2b  + blk * F_DIM * F_DIM;
    const bf16_t* M1 = mw1b + blk * F_DIM * 2 * F_DIM;
    const bf16_t* M2 = mw2b + blk * 2 * F_DIM * F_DIM;
    const float*  B1 = b1g + blk * F_DIM;
    const float*  B2 = b2g + blk * 2 * F_DIM;
    const int n0 = w * 16;

    // v2 = V @ W2^T   (M=96, N=128, K=128)
    for (int mt = 0; mt < (3 * TM) / 16; ++mt) {
      v8f acc = {};
#pragma unroll
      for (int kt = 0; kt < F_DIM / 32; ++kt) {
        v16bf a = load_a_frag(vA, F_DIM, mt * 16, kt * 32, lane);
        v16bf b = load_b_frag(W2, F_DIM, n0, kt * 32, lane);
        acc = wmma_bf16(a, b, acc);
      }
      store_d_bf16(vT, F_DIM, mt * 16, n0, lane, acc);
    }
    __syncthreads();

    // |v2| over Cartesian axis
    for (int idx = tid; idx < TM * F_DIM; idx += NTHREADS) {
      const int m = idx / F_DIM, g = idx % F_DIM;
      const float a0 = (float)vT[(3 * m + 0) * F_DIM + g];
      const float a1 = (float)vT[(3 * m + 1) * F_DIM + g];
      const float a2 = (float)vT[(3 * m + 2) * F_DIM + g];
      v2nB[idx] = (bf16_t)sqrtf(a0 * a0 + a1 * a1 + a2 * a2);
    }
    __syncthreads();

    // v1 = V @ W1^T (overwrites vT)
    for (int mt = 0; mt < (3 * TM) / 16; ++mt) {
      v8f acc = {};
#pragma unroll
      for (int kt = 0; kt < F_DIM / 32; ++kt) {
        v16bf a = load_a_frag(vA, F_DIM, mt * 16, kt * 32, lane);
        v16bf b = load_b_frag(W1, F_DIM, n0, kt * 32, lane);
        acc = wmma_bf16(a, b, acc);
      }
      store_d_bf16(vT, F_DIM, mt * 16, n0, lane, acc);
    }
    // MLP1: h = silu([s | |v2|] @ M1^T + b1)   (M=32, N=128, K=256)
    {
      const float bias = B1[n0 + (lane & 15)];
      for (int mt = 0; mt < TM / 16; ++mt) {
        v8f acc = {bias, bias, bias, bias, bias, bias, bias, bias};
#pragma unroll
        for (int kt = 0; kt < (2 * F_DIM) / 32; ++kt) {
          const bf16_t* Asrc = (kt < 4) ? sA : v2nB;
          const int kb = (kt < 4) ? kt * 32 : (kt - 4) * 32;
          v16bf a = load_a_frag(Asrc, F_DIM, mt * 16, kb, lane);
          v16bf b = load_b_frag(M1, 2 * F_DIM, n0, kt * 32, lane);
          acc = wmma_bf16(a, b, acc);
        }
        const int n  = n0 + (lane & 15);
        const int m0 = mt * 16 + ((lane >> 4) << 3);
#pragma unroll
        for (int r = 0; r < 8; ++r) {
          const float xv = acc[r];
          hA[(m0 + r) * F_DIM + n] = (bf16_t)(xv / (1.f + __expf(-xv)));
        }
      }
    }
    __syncthreads();

    // MLP2: h2 = h @ M2^T + b2 ; split into s_out / gate  (M=32, N=256, K=128)
    for (int half = 0; half < 2; ++half) {
      const int nn0 = n0 + half * 128;
      const float bias = B2[nn0 + (lane & 15)];
      for (int mt = 0; mt < TM / 16; ++mt) {
        v8f acc = {bias, bias, bias, bias, bias, bias, bias, bias};
#pragma unroll
        for (int kt = 0; kt < F_DIM / 32; ++kt) {
          v16bf a = load_a_frag(hA, F_DIM, mt * 16, kt * 32, lane);
          v16bf b = load_b_frag(M2, F_DIM, nn0, kt * 32, lane);
          acc = wmma_bf16(a, b, acc);
        }
        if (half == 0) store_d_bf16(sA,    F_DIM, mt * 16, nn0,       lane, acc);
        else           store_d_bf16(gateB, F_DIM, mt * 16, nn0 - 128, lane, acc);
      }
    }
    __syncthreads();

    // v_new = v1 * gate
    for (int idx = tid; idx < 3 * TM * F_DIM; idx += NTHREADS) {
      const int row = idx / F_DIM, g = idx % F_DIM;
      const int m = row / 3;
      vA[idx] = (bf16_t)((float)vT[idx] * (float)gateB[m * F_DIM + g]);
    }
    __syncthreads();
  }

  if (tid < TM) {
    const int node = node0 + tid;
    if (node < nnodes) {
      float acc = outb[0];
#pragma unroll 4
      for (int g = 0; g < F_DIM; ++g) acc += (float)sA[tid * F_DIM + g] * outw[g];
      sfin[node] = acc;
    }
  }
}

// ---------------------------------------------------------------------------
__global__ __launch_bounds__(256)
void seg_reduce(const float* __restrict__ mask, const float* __restrict__ sfin,
                float* __restrict__ y, int nnodes) {
  const int m = blockIdx.x;
  const float* row = mask + (size_t)m * nnodes;
  float acc = 0.f;
  const int n4 = nnodes & ~3;
  for (int n = threadIdx.x * 4; n < n4; n += 256 * 4) {
    const float4 a = *(const float4*)(row + n);
    const float4 b = *(const float4*)(sfin + n);
    acc += a.x * b.x + a.y * b.y + a.z * b.z + a.w * b.w;
  }
  for (int n = n4 + threadIdx.x; n < nnodes; n += 256) acc += row[n] * sfin[n];

  __shared__ float red[256];
  red[threadIdx.x] = acc;
  __syncthreads();
  for (int s = 128; s > 0; s >>= 1) {
    if (threadIdx.x < s) red[threadIdx.x] += red[threadIdx.x + s];
    __syncthreads();
  }
  if (threadIdx.x == 0) y[m] = red[0];
}

// ---------------------------------------------------------------------------
extern "C" void kernel_launch(void* const* d_in, const int* in_sizes, int n_in,
                              void* d_out, int out_size, void* d_ws, size_t ws_size,
                              hipStream_t stream) {
  const float* s    = (const float*)d_in[0];
  const float* v    = (const float*)d_in[1];
  // d_in[2] = r (unused by the reference)
  const float* bm   = (const float*)d_in[3];
  const float* w1   = (const float*)d_in[4];
  const float* w2   = (const float*)d_in[5];
  const float* mw1  = (const float*)d_in[6];
  const float* mb1  = (const float*)d_in[7];
  const float* mw2  = (const float*)d_in[8];
  const float* mb2  = (const float*)d_in[9];
  const float* outw = (const float*)d_in[10];
  const float* outb = (const float*)d_in[11];
  float* y = (float*)d_out;

  const int nnodes = in_sizes[0] / F_DIM;   // 50000
  const int nmol   = out_size;              // 256

  bf16_t* wb   = (bf16_t*)d_ws;             // 196608 bf16
  float*  sfin = (float*)((char*)d_ws + 196608 * sizeof(bf16_t));

  cvt_weights_bf16<<<(196608 + 255) / 256, 256, 0, stream>>>(w1, w2, mw1, mw2, wb);
  gated_equiv_fused<<<(nnodes + TM - 1) / TM, NTHREADS, 0, stream>>>(
      s, v, wb, mb1, mb2, outw, outb, sfin, nnodes);
  seg_reduce<<<nmol, 256, 0, stream>>>(bm, sfin, y, nnodes);
}
